// KnowledgeRetrievalNetwork_9234179687023
// MI455X (gfx1250) — matrix operand order repaired and checked by reading, not compiled, not run on hardware
//
#include <hip/hip_runtime.h>
#include <hip/hip_bf16.h>
#include <stdint.h>

// ---------------------------------------------------------------------------
// Problem constants (match reference setup_inputs exactly)
// ---------------------------------------------------------------------------
constexpr int kB = 4096;   // batch
constexpr int kD = 2048;   // input dim
constexpr int kH = 2048;   // hidden dim
constexpr int kE = 8;      // experts
constexpr float kSlope = 5.0f;

typedef __attribute__((ext_vector_type(16))) _Float16 v16h;
typedef __attribute__((ext_vector_type(8)))  float    v8f;

// ---------------------------------------------------------------------------
// Stateless counter-based RNG (deterministic across graph replays)
// ---------------------------------------------------------------------------
__device__ __forceinline__ uint32_t hash32(uint32_t x) {
  x ^= x >> 16; x *= 0x7feb352dU;
  x ^= x >> 15; x *= 0x846ca68bU;
  x ^= x >> 16;
  return x;
}
__device__ __forceinline__ float rand01(uint32_t idx, uint32_t salt) {
  return (float)(hash32(idx + salt * 0x9E3779B9U) >> 8) * (1.0f / 16777216.0f);
}

// ---------------------------------------------------------------------------
// CDNA5 async copy: global -> LDS, 16B per lane, tracked by ASYNCcnt.
// VDST operand carries the per-lane LDS byte offset (low 32 bits of a generic
// pointer into __shared__ == LDS offset per the aperture rules).
// Async loads complete in order, so s_wait_asynccnt N can leave the N most
// recently issued requests in flight (3-deep software pipeline below).
// ---------------------------------------------------------------------------
__device__ __forceinline__ void async_copy_b128(const _Float16* gptr,
                                                const _Float16* lptr) {
  uint32_t lds = (uint32_t)(uintptr_t)lptr;
  asm volatile("global_load_async_to_lds_b128 %0, %1, off"
               :: "v"(lds), "v"(gptr) : "memory");
}
__device__ __forceinline__ void async_wait_le3() {
  asm volatile("s_wait_asynccnt 3" ::: "memory");
}
__device__ __forceinline__ void async_wait_le0() {
  asm volatile("s_wait_asynccnt 0" ::: "memory");
}

// ---------------------------------------------------------------------------
// Kernel 0: fp32 -> f16 weight down-convert (activations are exact {0,1}
// spikes, so f16 WMMA is the right precision / throughput path on CDNA5)
// ---------------------------------------------------------------------------
__global__ __launch_bounds__(256) void cvt_f32_f16(const float* __restrict__ src,
                                                   _Float16* __restrict__ dst,
                                                   int n4) {
  int i = blockIdx.x * 256 + threadIdx.x;
  if (i >= n4) return;
  float4 v = ((const float4*)src)[i];
  union { float2 f2; _Float16 h[4]; } u;
  u.h[0] = (_Float16)v.x; u.h[1] = (_Float16)v.y;
  u.h[2] = (_Float16)v.z; u.h[3] = (_Float16)v.w;
  ((float2*)dst)[i] = u.f2;
}

// ---------------------------------------------------------------------------
// Kernel 1: spike-encode query (qs, f16) + gate softmax (one wave32 per row)
// ---------------------------------------------------------------------------
__global__ __launch_bounds__(256) void qs_gate_kernel(const float* __restrict__ q,
                                                      const float* __restrict__ Wg,
                                                      const float* __restrict__ bg,
                                                      _Float16* __restrict__ qs,
                                                      float* __restrict__ gate) {
  const int wid  = threadIdx.x >> 5;
  const int lane = threadIdx.x & 31;
  const int b    = blockIdx.x * 8 + wid;

  float acc[kE];
#pragma unroll
  for (int e = 0; e < kE; ++e) acc[e] = 0.0f;

  const float* qrow = q + (size_t)b * kD;
  for (int d = lane; d < kD; d += 32) {
    float p = fminf(fmaxf(qrow[d], 0.0f), 1.0f);
    float u = rand01((uint32_t)(b * kD + d), 0x2545F491u);
    float s = (u < p) ? 1.0f : 0.0f;
    qs[(size_t)b * kD + d] = (_Float16)s;
    if (s != 0.0f) {
#pragma unroll
      for (int e = 0; e < kE; ++e) acc[e] += Wg[e * kD + d];
    }
  }
  // wave32 butterfly reduction of 8 gate logits
#pragma unroll
  for (int e = 0; e < kE; ++e) {
#pragma unroll
    for (int off = 16; off; off >>= 1) acc[e] += __shfl_xor(acc[e], off, 32);
  }
  if (lane == 0) {
    float mx = -1e30f;
#pragma unroll
    for (int e = 0; e < kE; ++e) { acc[e] += bg[e]; mx = fmaxf(mx, acc[e]); }
    float sum = 0.0f;
#pragma unroll
    for (int e = 0; e < kE; ++e) { acc[e] = __expf(acc[e] - mx); sum += acc[e]; }
    float inv = 1.0f / sum;
#pragma unroll
    for (int e = 0; e < kE; ++e) gate[b * kE + e] = acc[e] * inv;
  }
}

// ---------------------------------------------------------------------------
// GEMM tiling: 128x64 block tile, K-step 32, 8 waves as 4(M) x 2(N); each
// wave owns a 32x32 register tile -> 4 v_wmma_f32_16x16x32_f16 per K-step
// with full A/B fragment reuse (2 LDS b128 loads per WMMA).
// Staging: GLOBAL_LOAD_ASYNC_TO_LDS_B128 into a 3-deep ring of LDS buffers.
// At step kt we issue tile kt+2 and wait only for tile kt+1 (asynccnt<=3,
// in-order completion), so global->LDS latency is hidden by a full K-step
// of WMMA work. Rows padded to 40 halfs to break ds_load_b128 bank conflicts.
// ---------------------------------------------------------------------------
constexpr int kBM  = 128;
constexpr int kBN  = 64;
constexpr int kBK  = 32;
constexpr int kPAD = 40;
constexpr int kNB  = 3;    // LDS ring depth

// Kernel 2: lin1 = qs @ W1[e]^T, fused spike layer 1 -> s1 (f16, {0,1})
__global__ __launch_bounds__(256) void gemm1_spike(const _Float16* __restrict__ qs,
                                                   const _Float16* __restrict__ W1h,
                                                   const float* __restrict__ th1,
                                                   _Float16* __restrict__ s1) {
  __shared__ _Float16 As[kNB][kBM][kPAD];
  __shared__ _Float16 Bs[kNB][kBN][kPAD];

  const int    e   = blockIdx.z;
  const size_t bm0 = (size_t)blockIdx.x * kBM;
  const size_t hn0 = (size_t)blockIdx.y * kBN;
  const _Float16* Be = W1h + (size_t)e * kH * kD;   // [H,D] row-major (K-major)

  const int tid  = threadIdx.x;
  const int lane = tid & 31, ll = lane & 15, lh = lane >> 4;
  const int wid  = tid >> 5, wm = wid & 3, wn = wid >> 2;

  // staging assignment: A 128x32 halfs (2 b128/thread), B 64x32 (1 b128/thread)
  const int rowA = tid >> 1, colA = (tid & 1) * 16;   // chunks colA, colA+8
  const int rowB = tid >> 2, colB = (tid & 3) * 8;

  const _Float16* aRow = qs + (bm0 + rowA) * kD + colA;
  const _Float16* bRow = Be + (hn0 + rowB) * kD + colB;

  auto stage = [&](int kt, int bufw) {
    const size_t ko = (size_t)kt * kBK;
    async_copy_b128(aRow + ko,     &As[bufw][rowA][colA]);
    async_copy_b128(aRow + ko + 8, &As[bufw][rowA][colA + 8]);
    async_copy_b128(bRow + ko,     &Bs[bufw][rowB][colB]);
  };

  v8f acc[4] = {{}, {}, {}, {}};

  // prologue: tiles 0 and 1 in flight; wait for tile 0 only
  stage(0, 0);
  stage(1, 1);
  async_wait_le3();
  __syncthreads();

  constexpr int NT = kD / kBK;   // 64 K-steps
  int bufR = 0;
  for (int kt = 0; kt < NT; ++kt) {
    const int bufW = (bufR >= 1) ? bufR - 1 : kNB - 1;   // == (kt+2)%3
    if (kt + 2 < NT) {
      stage(kt + 2, bufW);
      if (kt + 6 < NT) {
        __builtin_prefetch(aRow + (size_t)(kt + 6) * kBK, 0, 0);  // global_prefetch_b8
        __builtin_prefetch(bRow + (size_t)(kt + 6) * kBK, 0, 0);
      }
    }
    // A fragment: lane row M = ll, K = {lh*8..+7, 16+lh*8..+7}
    const _Float16* a0p = &As[bufR][wm * 32 + ll][lh * 8];
    const _Float16* a1p = &As[bufR][wm * 32 + 16 + ll][lh * 8];
    // B fragment: lane column N = ll, K = {lh*8..+7, 16+lh*8..+7}
    const _Float16* b0p = &Bs[bufR][wn * 32 + ll][lh * 8];
    const _Float16* b1p = &Bs[bufR][wn * 32 + 16 + ll][lh * 8];
    v16h af0, af1, bf0, bf1;
    ((float4*)&af0)[0] = *(const float4*)(a0p);
    ((float4*)&af0)[1] = *(const float4*)(a0p + 16);
    ((float4*)&af1)[0] = *(const float4*)(a1p);
    ((float4*)&af1)[1] = *(const float4*)(a1p + 16);
    ((float4*)&bf0)[0] = *(const float4*)(b0p);
    ((float4*)&bf0)[1] = *(const float4*)(b0p + 16);
    ((float4*)&bf1)[0] = *(const float4*)(b1p);
    ((float4*)&bf1)[1] = *(const float4*)(b1p + 16);

    acc[0] = __builtin_amdgcn_wmma_f32_16x16x32_f16(false, af0, false, bf0, (short)0, acc[0], false, false);
    acc[1] = __builtin_amdgcn_wmma_f32_16x16x32_f16(false, af0, false, bf1, (short)0, acc[1], false, false);
    acc[2] = __builtin_amdgcn_wmma_f32_16x16x32_f16(false, af1, false, bf0, (short)0, acc[2], false, false);
    acc[3] = __builtin_amdgcn_wmma_f32_16x16x32_f16(false, af1, false, bf1, (short)0, acc[3], false, false);

    if (kt + 1 < NT) {
      if (kt + 2 < NT) async_wait_le3();   // tile kt+1 done; kt+2 in flight
      else             async_wait_le0();   // nothing new issued: drain
      __syncthreads();
    }
    bufR = (bufR + 1 == kNB) ? 0 : bufR + 1;
  }

  // fused spike layer 1: bernoulli(sigmoid(5*(lin1 - th1)))
#pragma unroll
  for (int nt = 0; nt < 2; ++nt) {
    const size_t h  = hn0 + wn * 32 + nt * 16 + ll;
    const float  th = th1[e * kH + h];
#pragma unroll
    for (int mt = 0; mt < 2; ++mt) {
      const v8f c = acc[mt * 2 + nt];
#pragma unroll
      for (int r = 0; r < 8; ++r) {
        const int    m  = wm * 32 + mt * 16 + lh * 8 + r;
        const size_t b  = bm0 + m;
        const float  dv = c[r] - th;
        const float  p  = 1.0f / (1.0f + __expf(-kSlope * dv));
        const size_t gi = ((size_t)e * kB + b) * kH + h;
        const float  u  = rand01((uint32_t)gi, 0xB5297A4Du);
        s1[gi] = (_Float16)((u < p) ? 1.0f : 0.0f);
      }
    }
  }
}

// Kernel 3: for each expert: lin2 = s1[e] @ W2[e]^T, spike layer 2, then
// gate-weighted combine accumulated in registers -> out [B,H] f32
__global__ __launch_bounds__(256) void gemm2_spike_gate(const _Float16* __restrict__ s1,
                                                        const _Float16* __restrict__ W2h,
                                                        const float* __restrict__ th2,
                                                        const float* __restrict__ gate,
                                                        float* __restrict__ out) {
  __shared__ _Float16 As[kNB][kBM][kPAD];
  __shared__ _Float16 Bs[kNB][kBN][kPAD];

  const size_t bm0 = (size_t)blockIdx.x * kBM;
  const size_t hn0 = (size_t)blockIdx.y * kBN;

  const int tid  = threadIdx.x;
  const int lane = tid & 31, ll = lane & 15, lh = lane >> 4;
  const int wid  = tid >> 5, wm = wid & 3, wn = wid >> 2;

  const int rowA = tid >> 1, colA = (tid & 1) * 16;
  const int rowB = tid >> 2, colB = (tid & 3) * 8;

  v8f facc[4] = {{}, {}, {}, {}};   // gate-weighted final accumulators

  for (int e = 0; e < kE; ++e) {
    const _Float16* Ae = s1  + (size_t)e * kB * kH;   // [B,H] f16 spikes
    const _Float16* Be = W2h + (size_t)e * kH * kH;   // [K_out=H, H] K-major
    const _Float16* aRow = Ae + (bm0 + rowA) * kH + colA;
    const _Float16* bRow = Be + (hn0 + rowB) * kH + colB;

    auto stage = [&](int kt, int bufw) {
      const size_t ko = (size_t)kt * kBK;
      async_copy_b128(aRow + ko,     &As[bufw][rowA][colA]);
      async_copy_b128(aRow + ko + 8, &As[bufw][rowA][colA + 8]);
      async_copy_b128(bRow + ko,     &Bs[bufw][rowB][colB]);
    };

    v8f acc[4] = {{}, {}, {}, {}};

    // ring prologue (end-of-expert barrier guarantees buffers are free)
    stage(0, 0);
    stage(1, 1);
    async_wait_le3();
    __syncthreads();

    constexpr int NT = kH / kBK;   // 64
    int bufR = 0;
    for (int kt = 0; kt < NT; ++kt) {
      const int bufW = (bufR >= 1) ? bufR - 1 : kNB - 1;
      if (kt + 2 < NT) {
        stage(kt + 2, bufW);
        if (kt + 6 < NT) {
          __builtin_prefetch(aRow + (size_t)(kt + 6) * kBK, 0, 0);
          __builtin_prefetch(bRow + (size_t)(kt + 6) * kBK, 0, 0);
        }
      }
      const _Float16* a0p = &As[bufR][wm * 32 + ll][lh * 8];
      const _Float16* a1p = &As[bufR][wm * 32 + 16 + ll][lh * 8];
      const _Float16* b0p = &Bs[bufR][wn * 32 + ll][lh * 8];
      const _Float16* b1p = &Bs[bufR][wn * 32 + 16 + ll][lh * 8];
      v16h af0, af1, bf0, bf1;
      ((float4*)&af0)[0] = *(const float4*)(a0p);
      ((float4*)&af0)[1] = *(const float4*)(a0p + 16);
      ((float4*)&af1)[0] = *(const float4*)(a1p);
      ((float4*)&af1)[1] = *(const float4*)(a1p + 16);
      ((float4*)&bf0)[0] = *(const float4*)(b0p);
      ((float4*)&bf0)[1] = *(const float4*)(b0p + 16);
      ((float4*)&bf1)[0] = *(const float4*)(b1p);
      ((float4*)&bf1)[1] = *(const float4*)(b1p + 16);

      acc[0] = __builtin_amdgcn_wmma_f32_16x16x32_f16(false, af0, false, bf0, (short)0, acc[0], false, false);
      acc[1] = __builtin_amdgcn_wmma_f32_16x16x32_f16(false, af0, false, bf1, (short)0, acc[1], false, false);
      acc[2] = __builtin_amdgcn_wmma_f32_16x16x32_f16(false, af1, false, bf0, (short)0, acc[2], false, false);
      acc[3] = __builtin_amdgcn_wmma_f32_16x16x32_f16(false, af1, false, bf1, (short)0, acc[3], false, false);

      if (kt + 1 < NT) {
        if (kt + 2 < NT) async_wait_le3();
        else             async_wait_le0();
        __syncthreads();
      }
      bufR = (bufR + 1 == kNB) ? 0 : bufR + 1;
    }

    // fused spike layer 2 + gate-weighted combine
#pragma unroll
    for (int nt = 0; nt < 2; ++nt) {
      const size_t h  = hn0 + wn * 32 + nt * 16 + ll;
      const float  th = th2[e * kH + h];
#pragma unroll
      for (int mt = 0; mt < 2; ++mt) {
        const v8f c = acc[mt * 2 + nt];
#pragma unroll
        for (int r = 0; r < 8; ++r) {
          const int    m  = wm * 32 + mt * 16 + lh * 8 + r;
          const size_t b  = bm0 + m;
          const float  dv = c[r] - th;
          const float  p  = 1.0f / (1.0f + __expf(-kSlope * dv));
          const size_t gi = ((size_t)e * kB + b) * kH + h;
          const float  u  = rand01((uint32_t)gi, 0x68E31DA4u);
          const float  s  = (u < p) ? 1.0f : 0.0f;
          facc[mt * 2 + nt][r] += s * gate[b * kE + e];
        }
      }
    }
    __syncthreads();   // ring buffers must be quiesced before next expert
  }

  // write final [B,H] output (overwrites poison fully)
#pragma unroll
  for (int nt = 0; nt < 2; ++nt) {
    const size_t h = hn0 + wn * 32 + nt * 16 + ll;
#pragma unroll
    for (int mt = 0; mt < 2; ++mt) {
#pragma unroll
      for (int r = 0; r < 8; ++r) {
        const int    m = wm * 32 + mt * 16 + lh * 8 + r;
        const size_t b = bm0 + m;
        out[b * kH + h] = facc[mt * 2 + nt][r];
      }
    }
  }
}

// ---------------------------------------------------------------------------
// Launcher
// ---------------------------------------------------------------------------
extern "C" void kernel_launch(void* const* d_in, const int* in_sizes, int n_in,
                              void* d_out, int out_size, void* d_ws, size_t ws_size,
                              hipStream_t stream) {
  const float* q   = (const float*)d_in[0];   // [B,D]
  const float* W1  = (const float*)d_in[1];   // [E,H,D]
  const float* W2  = (const float*)d_in[2];   // [E,H,H]
  const float* th1 = (const float*)d_in[3];   // [E,H]
  const float* th2 = (const float*)d_in[4];   // [E,H]
  const float* Wg  = (const float*)d_in[5];   // [E,D]
  const float* bg  = (const float*)d_in[6];   // [E]
  float* out = (float*)d_out;                 // [B,H]

  char* ws = (char*)d_ws;
  size_t off = 0;
  auto take = [&](size_t bytes) {
    void* p = ws + off;
    off = (off + bytes + 255) & ~(size_t)255;
    return p;
  };
  _Float16* qs   = (_Float16*)take((size_t)kB * kD * sizeof(_Float16));
  float*    gate = (float*)   take((size_t)kB * kE * sizeof(float));
  _Float16* W1h  = (_Float16*)take((size_t)kE * kH * kD * sizeof(_Float16));
  _Float16* W2h  = (_Float16*)take((size_t)kE * kH * kH * sizeof(_Float16));
  _Float16* s1   = (_Float16*)take((size_t)kE * kB * kH * sizeof(_Float16));
  (void)ws_size; (void)in_sizes; (void)n_in; (void)out_size;

  const int n4 = kE * kH * kD / 4;   // == kE*kH*kH/4 (D == H)
  cvt_f32_f16<<<n4 / 256, 256, 0, stream>>>(W1, W1h, n4);
  cvt_f32_f16<<<n4 / 256, 256, 0, stream>>>(W2, W2h, n4);

  qs_gate_kernel<<<kB / 8, 256, 0, stream>>>(q, Wg, bg, qs, gate);

  gemm1_spike<<<dim3(kB / kBM, kH / kBN, kE), 256, 0, stream>>>(qs, W1h, th1, s1);

  gemm2_spike_gate<<<dim3(kB / kBM, kH / kBN), 256, 0, stream>>>(s1, W2h, th2, gate, out);
}
